// PerformerAttention_68822555951223
// MI455X (gfx1250) — matrix operand order repaired and checked by reading, not compiled, not run on hardware
//
#include <hip/hip_runtime.h>

#define BN 16384   // B*N tokens
#define CC 768     // channels
#define C3 2304    // 3*C
#define HH 12      // heads
#define DD 64      // head dim
#define NN 4096    // seq len
#define MF 256     // performer features
#define NHEADS 48  // B*H
#define NSPLIT 8   // n-dimension split for context partials
#define NSROWS (NN / NSPLIT)  // 512 rows per partial block
#define EPS 1e-3f
#define QKNORM 0.35355339059327373f  // 64^-0.25

typedef _Float16 half8 __attribute__((ext_vector_type(8)));
typedef _Float16 v16h  __attribute__((ext_vector_type(16)));
typedef float    v8f   __attribute__((ext_vector_type(8)));

__device__ __forceinline__ v8f wmma16(v16h a, v16h b, v8f c) {
  return __builtin_amdgcn_wmma_f32_16x16x32_f16(false, a, false, b, (short)0, c, false, false);
}

// A fragment (16x32 f16, M x K): lane holds row = lane&15.
// halves 0..7  : K = kb + (lane>=16 ? 8 : 0) + 0..7
// halves 8..15 : K = kb + 16 + (lane>=16 ? 8 : 0) + 0..7
__device__ __forceinline__ v16h load_afrag(const _Float16* base, int ld, int row, int kb, int lane) {
  const _Float16* p = base + (size_t)row * ld + kb + ((lane & 16) ? 8 : 0);
  half8 lo = *(const half8*)(p);
  half8 hi = *(const half8*)(p + 16);
  return __builtin_shufflevector(lo, hi, 0,1,2,3,4,5,6,7,8,9,10,11,12,13,14,15);
}

// B fragment (32x16 f16, K x N): lane holds col = lane&15.
// halves 0..7  : K = kb + (lane>=16 ? 8 : 0) + 0..7
// halves 8..15 : K = kb + 16 + (lane>=16 ? 8 : 0) + 0..7
__device__ __forceinline__ v16h load_bfrag(const _Float16* base, int ld, int col, int kb, int lane) {
  const _Float16* p = base + (size_t)col * ld + kb + ((lane & 16) ? 8 : 0);
  half8 lo = *(const half8*)(p);
  half8 hi = *(const half8*)(p + 16);
  return __builtin_shufflevector(lo, hi, 0,1,2,3,4,5,6,7,8,9,10,11,12,13,14,15);
}

__global__ __launch_bounds__(256) void cvt_kernel(const float* __restrict__ src,
                                                  _Float16* __restrict__ dst,
                                                  int n, float scale) {
  int i = blockIdx.x * 256 + threadIdx.x;
  if (i < n) dst[i] = (_Float16)(src[i] * scale);
}

// ---------------------------------------------------------------------------
// Kernel 1: qkv = x @ w_qkv^T + b_qkv, scattered to per-head f16 q/k/v.
// grid (C3/128=18, BN/128=128), 256 threads = 8 waves (2 x 4 wave grid).
// ---------------------------------------------------------------------------
__global__ __launch_bounds__(256) void gemm_qkv_kernel(
    const _Float16* __restrict__ xh, const _Float16* __restrict__ wh,
    const float* __restrict__ bias,
    _Float16* __restrict__ qh, _Float16* __restrict__ kh, _Float16* __restrict__ vh) {
  const int lane = threadIdx.x & 31;
  const int wave = threadIdx.x >> 5;
  const int M0 = blockIdx.y * 128 + (wave >> 2) * 64;
  const int N0 = blockIdx.x * 128 + (wave & 3) * 32;

  v8f acc[4][2];
#pragma unroll
  for (int r = 0; r < 4; ++r)
#pragma unroll
    for (int c = 0; c < 2; ++c) acc[r][c] = (v8f){0.f,0.f,0.f,0.f,0.f,0.f,0.f,0.f};

  for (int kb = 0; kb < CC; kb += 32) {
    v16h a[4], b[2];
#pragma unroll
    for (int r = 0; r < 4; ++r) a[r] = load_afrag(xh, CC, M0 + r * 16 + (lane & 15), kb, lane);
#pragma unroll
    for (int c = 0; c < 2; ++c) b[c] = load_bfrag(wh, CC, N0 + c * 16 + (lane & 15), kb, lane);
#pragma unroll
    for (int r = 0; r < 4; ++r)
#pragma unroll
      for (int c = 0; c < 2; ++c) acc[r][c] = wmma16(a[r], b[c], acc[r][c]);
  }

#pragma unroll
  for (int r = 0; r < 4; ++r)
#pragma unroll
    for (int c = 0; c < 2; ++c) {
      const int j = N0 + c * 16 + (lane & 15);
      const float bj = bias[j];
      const int s = j / CC;          // 0=q 1=k 2=v
      const int rem = j - s * CC;
      const int h = rem >> 6;
      const int dd = rem & 63;
      _Float16* dst = (s == 0) ? qh : ((s == 1) ? kh : vh);
#pragma unroll
      for (int i = 0; i < 8; ++i) {
        const int gm = M0 + r * 16 + i + ((lane & 16) ? 8 : 0);
        const int bb = gm >> 12;             // / NN
        const int n  = gm & (NN - 1);
        dst[(((size_t)bb * HH + h) * NN + n) * DD + dd] = (_Float16)(acc[r][c][i] + bj);
      }
    }
}

// ---------------------------------------------------------------------------
// Kernel 2a: partial context per (n-slice, head).
// Each block: 512 rows of n. kp = relu(k @ (norm*P)^T)+eps via WMMA,
// staged transposed in LDS, context partial accumulated in registers.
// Partial output (f32): ctxP[head][ns][r][m], r in 0..64 (64 = k_sum row).
// k_sum combined deterministically through unique LDS slots (no atomics).
// grid (NSPLIT=8, 48), 256 threads.
// ---------------------------------------------------------------------------
__global__ __launch_bounds__(256) void ctx_partial_kernel(
    const _Float16* __restrict__ kh, const _Float16* __restrict__ vh,
    const _Float16* __restrict__ ph, float* __restrict__ ctxP) {
  __shared__ alignas(64) _Float16 kpT[256][32];  // [m][n_local]
  __shared__ alignas(64) _Float16 vT[64][32];    // [d][n_local]
  __shared__ float ksp[8][2][64];                // [wave][lane-half][m&63]

  const int ns = blockIdx.x;
  const int bh = blockIdx.y;
  const _Float16* khead = kh + (size_t)bh * NN * DD;
  const _Float16* vhead = vh + (size_t)bh * NN * DD;
  float* ctxp = ctxP + ((size_t)bh * NSPLIT + ns) * 65 * 256;

  const int lane = threadIdx.x & 31;
  const int wave = threadIdx.x >> 5;

  // gemm1 split: wave -> n-subtile rt (0..1), m-tiles mtbase..mtbase+3
  const int rt = wave & 1;
  const int mtbase = (wave >> 1) * 4;
  // v staging indices
  const int vn = threadIdx.x >> 3;        // 0..31
  const int vd = (threadIdx.x & 7) * 8;   // 0,8,..,56

  // hoisted projection B fragments
  v16h bq0[4], bq1[4];
#pragma unroll
  for (int c = 0; c < 4; ++c) {
    const int m = (mtbase + c) * 16 + (lane & 15);
    bq0[c] = load_bfrag(ph, DD, m, 0, lane);
    bq1[c] = load_bfrag(ph, DD, m, 32, lane);
  }

  // context accumulators: wave owns m-tiles {2w,2w+1} x d-tiles 0..3
  v8f cacc[2][4];
#pragma unroll
  for (int mi = 0; mi < 2; ++mi)
#pragma unroll
    for (int dt = 0; dt < 4; ++dt) cacc[mi][dt] = (v8f){0.f,0.f,0.f,0.f,0.f,0.f,0.f,0.f};

  float ksacc[4] = {0.f, 0.f, 0.f, 0.f};

  const int nbeg = ns * NSROWS;
  for (int nc = nbeg; nc < nbeg + NSROWS; nc += 32) {
    // ---- kp chunk via WMMA ----
    const int arow = nc + rt * 16 + (lane & 15);
    v16h a0 = load_afrag(khead, DD, arow, 0, lane);
    v16h a1 = load_afrag(khead, DD, arow, 32, lane);
#pragma unroll
    for (int c = 0; c < 4; ++c) {
      v8f acc = (v8f){0.f,0.f,0.f,0.f,0.f,0.f,0.f,0.f};
      acc = wmma16(a0, bq0[c], acc);
      acc = wmma16(a1, bq1[c], acc);
      const int m = (mtbase + c) * 16 + (lane & 15);
      float s8 = 0.f;
      half8 pk;
#pragma unroll
      for (int i = 0; i < 8; ++i) {
        float v = acc[i] > 0.f ? acc[i] : 0.f;
        v += EPS;
        s8 += v;
        pk[i] = (_Float16)v;
      }
      ksacc[c] += s8;
      const int nl = rt * 16 + ((lane & 16) ? 8 : 0);
      *(half8*)&kpT[m][nl] = pk;
    }
    // ---- stage v^T ----
    {
      half8 vv = *(const half8*)(vhead + (size_t)(nc + vn) * DD + vd);
#pragma unroll
      for (int e = 0; e < 8; ++e) vT[vd + e][vn] = vv[e];
    }
    __syncthreads();
    // ---- context += kp^T @ v ----
#pragma unroll
    for (int mi = 0; mi < 2; ++mi) {
      v16h a = load_afrag(&kpT[0][0], 32, (wave * 2 + mi) * 16 + (lane & 15), 0, lane);
#pragma unroll
      for (int dt = 0; dt < 4; ++dt) {
        v16h b = load_bfrag(&vT[0][0], 32, dt * 16 + (lane & 15), 0, lane);
        cacc[mi][dt] = wmma16(a, b, cacc[mi][dt]);
      }
    }
    __syncthreads();
  }

  // partial context rows 0..63 (row = d, col = m), f32
#pragma unroll
  for (int mi = 0; mi < 2; ++mi)
#pragma unroll
    for (int dt = 0; dt < 4; ++dt) {
      const int dd = dt * 16 + (lane & 15);
      const int mbase = (wave * 2 + mi) * 16 + ((lane & 16) ? 8 : 0);
      *(v8f*)&ctxp[(size_t)dd * 256 + mbase] = cacc[mi][dt];
    }

  // deterministic k_sum combine: unique LDS slot per (wave, lane-half, m)
#pragma unroll
  for (int c = 0; c < 4; ++c)
    ksp[wave][lane >> 4][c * 16 + (lane & 15)] = ksacc[c];
  __syncthreads();
  {
    const int t = threadIdx.x;           // t == m (0..255)
    const int w0 = (t >> 6) * 2;         // wave pair covering this m
    const int li = t & 63;
    float s = ksp[w0][0][li] + ksp[w0][1][li] + ksp[w0 + 1][0][li] + ksp[w0 + 1][1][li];
    ctxp[64 * 256 + t] = s;              // row 64 = partial k_sum
  }
}

// ---------------------------------------------------------------------------
// Kernel 2b: reduce NSPLIT partials (fixed order) -> ctxT f16 [80][256]/head.
// Rows 0..63 = ctx^T, row 64 = k_sum, rows 65..79 = 0.
// grid (48), 256 threads.
// ---------------------------------------------------------------------------
__global__ __launch_bounds__(256) void ctx_reduce_kernel(
    const float* __restrict__ ctxP, _Float16* __restrict__ ctxT) {
  const int bh = blockIdx.x;
  const int t = threadIdx.x;
  const float* src = ctxP + (size_t)bh * NSPLIT * 65 * 256;
  _Float16* dst = ctxT + (size_t)bh * 80 * 256;
  for (int r = 0; r < 65; ++r) {
    float s = 0.f;
#pragma unroll
    for (int p = 0; p < NSPLIT; ++p) s += src[((size_t)p * 65 + r) * 256 + t];
    dst[r * 256 + t] = (_Float16)s;
  }
#pragma unroll
  for (int r = 65; r < 80; ++r) dst[r * 256 + t] = (_Float16)0.f;
}

// ---------------------------------------------------------------------------
// Kernel 3: per head, per 64-row n-tile:
//   qp = relu(q @ (norm*P)^T) + eps      (staged in LDS, f16)
//   [out | denom] = qp @ ctx_ext         (fused 64x80 GEMM, K=256)
//   midh[b,n,h*64+c] = out * (1/denom)
// grid (NN/64=64, 48), 256 threads.
// ---------------------------------------------------------------------------
__global__ __launch_bounds__(256) void outmid_kernel(
    const _Float16* __restrict__ qh, const _Float16* __restrict__ ph,
    const _Float16* __restrict__ ctxT, _Float16* __restrict__ midh) {
  __shared__ alignas(64) _Float16 qpL[64][256];
  __shared__ float dnm[64];

  const int bh = blockIdx.y;
  const int b = bh / HH;
  const int h = bh % HH;
  const int n0 = blockIdx.x * 64;
  const _Float16* qhead = qh + (size_t)bh * NN * DD;
  const _Float16* ctxh = ctxT + (size_t)bh * 80 * 256;

  const int lane = threadIdx.x & 31;
  const int wave = threadIdx.x >> 5;

  // ---- gemm1: qp tile (64 x 256). wave -> row slab (wave>>1), 8 m-tiles ----
  {
    const int rs = wave >> 1;               // 0..3, 16-row slab
    const int mt0 = (wave & 1) * 8;
    const int arow = n0 + rs * 16 + (lane & 15);
    v16h a0 = load_afrag(qhead, DD, arow, 0, lane);
    v16h a1 = load_afrag(qhead, DD, arow, 32, lane);
#pragma unroll
    for (int c = 0; c < 8; ++c) {
      const int m = (mt0 + c) * 16 + (lane & 15);
      v16h b0 = load_bfrag(ph, DD, m, 0, lane);
      v16h b1 = load_bfrag(ph, DD, m, 32, lane);
      v8f acc = (v8f){0.f,0.f,0.f,0.f,0.f,0.f,0.f,0.f};
      acc = wmma16(a0, b0, acc);
      acc = wmma16(a1, b1, acc);
      const int rl = rs * 16 + ((lane & 16) ? 8 : 0);
#pragma unroll
      for (int i = 0; i < 8; ++i) {
        float v = acc[i] > 0.f ? acc[i] : 0.f;
        qpL[rl + i][m] = (_Float16)(v + EPS);
      }
    }
  }
  __syncthreads();

  // ---- gemm2: qp (64x256) @ ctx_ext (256x80) ----
  // waves 0-3: col tiles {0,1,2}; waves 4-7: col tiles {3,4}. rt = wave&3.
  const int rt = wave & 3;
  const int nct = (wave < 4) ? 3 : 2;
  const int ct0 = (wave < 4) ? 0 : 3;
  v8f oacc[3];
#pragma unroll
  for (int c = 0; c < 3; ++c) oacc[c] = (v8f){0.f,0.f,0.f,0.f,0.f,0.f,0.f,0.f};

  for (int kb = 0; kb < 256; kb += 32) {
    v16h a = load_afrag(&qpL[0][0], 256, rt * 16 + (lane & 15), kb, lane);
    for (int c = 0; c < nct; ++c) {
      v16h bf = load_bfrag(ctxh, 256, (ct0 + c) * 16 + (lane & 15), kb, lane);
      oacc[c] = wmma16(a, bf, oacc[c]);
    }
  }

  // denominator lives in global col 64 = tile 4, col-lane 0 (waves 4-7)
  if (wave >= 4 && (lane & 15) == 0) {
    const int rbase = rt * 16 + ((lane & 16) ? 8 : 0);
#pragma unroll
    for (int i = 0; i < 8; ++i) dnm[rbase + i] = oacc[1][i];
  }
  __syncthreads();

  float rinv[8];
  {
    const int rbase = rt * 16 + ((lane & 16) ? 8 : 0);
#pragma unroll
    for (int i = 0; i < 8; ++i) rinv[i] = 1.0f / dnm[rbase + i];
  }

  const int nst = (wave < 4) ? 3 : 1;  // waves>=4 store only tile 3 (cols 48..63)
  for (int c = 0; c < nst; ++c) {
    const int cl = (ct0 + c) * 16 + (lane & 15);
#pragma unroll
    for (int i = 0; i < 8; ++i) {
      const int n = n0 + rt * 16 + i + ((lane & 16) ? 8 : 0);
      midh[((size_t)b * NN + n) * CC + h * DD + cl] = (_Float16)(oacc[c][i] * rinv[i]);
    }
  }
}

// ---------------------------------------------------------------------------
// Kernel 4: out = midh @ w_proj^T + b_proj (f32 output).
// grid (CC/128=6, BN/128=128), 256 threads.
// ---------------------------------------------------------------------------
__global__ __launch_bounds__(256) void gemm_proj_kernel(
    const _Float16* __restrict__ midh, const _Float16* __restrict__ wh,
    const float* __restrict__ bias, float* __restrict__ out) {
  const int lane = threadIdx.x & 31;
  const int wave = threadIdx.x >> 5;
  const int M0 = blockIdx.y * 128 + (wave >> 2) * 64;
  const int N0 = blockIdx.x * 128 + (wave & 3) * 32;

  v8f acc[4][2];
#pragma unroll
  for (int r = 0; r < 4; ++r)
#pragma unroll
    for (int c = 0; c < 2; ++c) acc[r][c] = (v8f){0.f,0.f,0.f,0.f,0.f,0.f,0.f,0.f};

  for (int kb = 0; kb < CC; kb += 32) {
    v16h a[4], b[2];
#pragma unroll
    for (int r = 0; r < 4; ++r) a[r] = load_afrag(midh, CC, M0 + r * 16 + (lane & 15), kb, lane);
#pragma unroll
    for (int c = 0; c < 2; ++c) b[c] = load_bfrag(wh, CC, N0 + c * 16 + (lane & 15), kb, lane);
#pragma unroll
    for (int r = 0; r < 4; ++r)
#pragma unroll
      for (int c = 0; c < 2; ++c) acc[r][c] = wmma16(a[r], b[c], acc[r][c]);
  }

#pragma unroll
  for (int r = 0; r < 4; ++r)
#pragma unroll
    for (int c = 0; c < 2; ++c) {
      const int j = N0 + c * 16 + (lane & 15);
      const float bj = bias[j];
#pragma unroll
      for (int i = 0; i < 8; ++i) {
        const int gm = M0 + r * 16 + i + ((lane & 16) ? 8 : 0);
        out[(size_t)gm * CC + j] = acc[r][c][i] + bj;
      }
    }
}

// ---------------------------------------------------------------------------
extern "C" void kernel_launch(void* const* d_in, const int* in_sizes, int n_in,
                              void* d_out, int out_size, void* d_ws, size_t ws_size,
                              hipStream_t stream) {
  const float* x     = (const float*)d_in[0];
  const float* w_qkv = (const float*)d_in[1];
  const float* b_qkv = (const float*)d_in[2];
  const float* w_prj = (const float*)d_in[3];
  const float* b_prj = (const float*)d_in[4];
  const float* proj  = (const float*)d_in[5];

  size_t off = 0;
  auto alloc = [&](size_t bytes) -> void* {
    void* p = (char*)d_ws + off;
    off += (bytes + 255) & ~(size_t)255;
    return p;
  };
  _Float16* xh   = (_Float16*)alloc((size_t)BN * CC * 2);
  _Float16* wqh  = (_Float16*)alloc((size_t)C3 * CC * 2);
  _Float16* wph  = (_Float16*)alloc((size_t)CC * CC * 2);
  _Float16* ph   = (_Float16*)alloc((size_t)MF * DD * 2);
  _Float16* qh   = (_Float16*)alloc((size_t)NHEADS * NN * DD * 2);
  _Float16* kh   = (_Float16*)alloc((size_t)NHEADS * NN * DD * 2);
  _Float16* vh   = (_Float16*)alloc((size_t)NHEADS * NN * DD * 2);
  _Float16* ctxT = (_Float16*)alloc((size_t)NHEADS * 80 * 256 * 2);
  float*    ctxP = (float*)   alloc((size_t)NHEADS * NSPLIT * 65 * 256 * 4);
  _Float16* midh = (_Float16*)alloc((size_t)BN * CC * 2);
  (void)ws_size; (void)n_in; (void)in_sizes; (void)out_size;

  // f32 -> f16 conversions (normalizer folded into projection)
  cvt_kernel<<<(BN * CC + 255) / 256, 256, 0, stream>>>(x, xh, BN * CC, 1.0f);
  cvt_kernel<<<(C3 * CC + 255) / 256, 256, 0, stream>>>(w_qkv, wqh, C3 * CC, 1.0f);
  cvt_kernel<<<(CC * CC + 255) / 256, 256, 0, stream>>>(w_prj, wph, CC * CC, 1.0f);
  cvt_kernel<<<(MF * DD + 255) / 256, 256, 0, stream>>>(proj, ph, MF * DD, QKNORM);

  gemm_qkv_kernel<<<dim3(C3 / 128, BN / 128), 256, 0, stream>>>(xh, wqh, b_qkv, qh, kh, vh);
  ctx_partial_kernel<<<dim3(NSPLIT, NHEADS), 256, 0, stream>>>(kh, vh, ph, ctxP);
  ctx_reduce_kernel<<<NHEADS, 256, 0, stream>>>(ctxP, ctxT);
  outmid_kernel<<<dim3(NN / 64, NHEADS), 256, 0, stream>>>(qh, ph, ctxT, midh);
  gemm_proj_kernel<<<dim3(CC / 128, BN / 128), 256, 0, stream>>>(midh, wph, b_prj, (float*)d_out);
}
